// AlignFormer_90812788506896
// MI455X (gfx1250) — compile-verified, matrix operand order checked by reading, not run on hardware
//
#include <hip/hip_runtime.h>

// ---------------- model constants ----------------
#define Dm    512
#define NAa   4096
#define NFf   512
#define Hh    8
#define DH    64
#define Cc    4
#define Kk    16
#define F1    1024   // FMID1
#define F2    512    // FMID2
#define NQKV  1536

typedef __attribute__((ext_vector_type(16))) __bf16 v16bf;
typedef __attribute__((ext_vector_type(8)))  float  v8f;

union Frag { v16bf v; unsigned u[8]; };

__device__ __forceinline__ unsigned short f2bf(float f) {
    union { float f; unsigned u; } x; x.f = f;
    unsigned u = x.u;
    u += 0x7FFFu + ((u >> 16) & 1u);   // round-to-nearest-even
    return (unsigned short)(u >> 16);
}

// packed-pair K index for 16-bit WMMA A/B fragments (ISA 7.12.2)
__device__ __forceinline__ int kpair(int v, int hf) {
    return (v < 4) ? (hf * 8 + 2 * v) : (16 + hf * 8 + 2 * (v - 4));
}

// ---------------- elementwise / prep kernels ----------------

// single-block scan of query_list -> atom->residue map
__global__ void build_resmap(const int* __restrict__ q, int* __restrict__ resmap) {
    __shared__ int sh[NFf];
    int t = threadIdx.x;
    sh[t] = q[t];
    __syncthreads();
    for (int off = 1; off < NFf; off <<= 1) {
        int v = (t >= off) ? sh[t - off] : 0;
        __syncthreads();
        sh[t] += v;
        __syncthreads();
    }
    int end = sh[t];
    int cnt = q[t];
    int start = end - cnt;
    for (int j = 0; j < cnt; ++j) resmap[start + j] = t;
}

// atom_in = out_atom + repeat(out_fasta)
__global__ void repeat_add(const float* __restrict__ atom, const float* __restrict__ fasta,
                           const int* __restrict__ resmap,
                           float* __restrict__ yf, unsigned short* __restrict__ yb) {
    int idx = blockIdx.x * blockDim.x + threadIdx.x;
    if (idx >= NAa * Dm) return;
    int a = idx >> 9, d = idx & 511;
    float v = atom[idx] + fasta[resmap[a] * Dm + d];
    yf[idx] = v;
    yb[idx] = f2bf(v);
}

__global__ void add2(const float* __restrict__ a, const float* __restrict__ b,
                     float* __restrict__ yf, unsigned short* __restrict__ yb, int n) {
    int idx = blockIdx.x * blockDim.x + threadIdx.x;
    if (idx >= n) return;
    float v = a[idx] + b[idx];
    yf[idx] = v;
    yb[idx] = f2bf(v);
}

__global__ void conv_bf16(const float* __restrict__ x, unsigned short* __restrict__ y, int n) {
    int idx = blockIdx.x * blockDim.x + threadIdx.x;
    if (idx < n) y[idx] = f2bf(x[idx]);
}

// weight [K,N] f32 -> [N,K] bf16 (transposed so B pairs along K are contiguous)
__global__ void tconv(const float* __restrict__ w, unsigned short* __restrict__ wt,
                      int K, int N) {
    int idx = blockIdx.x * blockDim.x + threadIdx.x;
    if (idx >= K * N) return;
    int k = idx / N, n = idx % N;
    wt[n * K + k] = f2bf(w[idx]);
}

// split qkv [S,1536] f32 -> q,k bf16 [S,512] and v^T bf16 [512,S]
__global__ void qkv_split(const float* __restrict__ qkv,
                          unsigned short* __restrict__ qb,
                          unsigned short* __restrict__ kb,
                          unsigned short* __restrict__ vtb, int S) {
    int idx = blockIdx.x * blockDim.x + threadIdx.x;
    if (idx >= S * Dm) return;
    int s = idx >> 9, d = idx & 511;
    const float* row = qkv + s * NQKV;
    qb[idx] = f2bf(row[d]);
    kb[idx] = f2bf(row[Dm + d]);
    vtb[d * S + s] = f2bf(row[2 * Dm + d]);
}

// y = LN(X (+R)) * g + b ; optional f32 and bf16 outputs. Row per block, 256 thr.
__global__ void ln512(const float* __restrict__ X, const float* __restrict__ R,
                      const float* __restrict__ g, const float* __restrict__ b,
                      float* __restrict__ Yf, unsigned short* __restrict__ Yb) {
    __shared__ float red[256];
    int row = blockIdx.x, t = threadIdx.x;
    float lv[2], s = 0.f;
#pragma unroll
    for (int i = 0; i < 2; ++i) {
        int c = t + i * 256;
        float x = X[row * Dm + c];
        if (R) x += R[row * Dm + c];
        lv[i] = x; s += x;
    }
    red[t] = s; __syncthreads();
    for (int o = 128; o > 0; o >>= 1) { if (t < o) red[t] += red[t + o]; __syncthreads(); }
    float mean = red[0] * (1.0f / Dm);
    __syncthreads();
    float s2 = 0.f;
#pragma unroll
    for (int i = 0; i < 2; ++i) { float d = lv[i] - mean; s2 += d * d; }
    red[t] = s2; __syncthreads();
    for (int o = 128; o > 0; o >>= 1) { if (t < o) red[t] += red[t + o]; __syncthreads(); }
    float rstd = rsqrtf(red[0] * (1.0f / Dm) + 1e-5f);
#pragma unroll
    for (int i = 0; i < 2; ++i) {
        int c = t + i * 256;
        float y = (lv[i] - mean) * rstd * g[c] + b[c];
        if (Yf) Yf[row * Dm + c] = y;
        if (Yb) Yb[row * Dm + c] = f2bf(y);
    }
}

// out[f,d] = sum_c max_k atom_trans[centroids[c,f,k], d]
__global__ void gather_max_sum(const float* __restrict__ at, const int* __restrict__ cen,
                               float* __restrict__ out) {
    int idx = blockIdx.x * blockDim.x + threadIdx.x;
    if (idx >= NFf * Dm) return;
    int f = idx >> 9, d = idx & 511;
    float acc = 0.f;
#pragma unroll
    for (int c = 0; c < Cc; ++c) {
        float mx = -3.0e38f;
        const int* row = cen + (c * NFf + f) * Kk;
#pragma unroll
        for (int kk = 0; kk < Kk; ++kk)
            mx = fmaxf(mx, at[row[kk] * Dm + d]);
        acc += mx;
    }
    out[idx] = acc;
}

// ---------------- WMMA GEMM ----------------
// C[M,N](f32) = A[M,K](bf16) @ Bt[N,K](bf16)^T  + bias ; optional relu / bf16 out
// block = 128 thr (4 waves), block tile 64x64, wave tile 32x32 (2x2 frags)
__global__ void gemm_wmma_bf16(const unsigned short* __restrict__ A,
                               const unsigned short* __restrict__ Bt,
                               const float* __restrict__ bias,
                               float* __restrict__ Cf, unsigned short* __restrict__ Cb,
                               int M, int N, int K, int relu) {
    int wave = threadIdx.x >> 5;
    int lane = threadIdx.x & 31;
    int ln = lane & 15, hf = lane >> 4;
    int m0 = blockIdx.y * 64 + (wave & 1) * 32;
    int n0 = blockIdx.x * 64 + (wave >> 1) * 32;

    v8f acc[2][2];
#pragma unroll
    for (int i = 0; i < 2; ++i)
#pragma unroll
        for (int j = 0; j < 2; ++j)
            acc[i][j] = (v8f){0.f, 0.f, 0.f, 0.f, 0.f, 0.f, 0.f, 0.f};

    for (int k0 = 0; k0 < K; k0 += 32) {
        Frag a[2], b[2];
#pragma unroll
        for (int mi = 0; mi < 2; ++mi) {
            const unsigned short* ap = A + (size_t)(m0 + mi * 16 + ln) * K + k0;
#pragma unroll
            for (int v = 0; v < 8; ++v)
                a[mi].u[v] = *(const unsigned*)(ap + kpair(v, hf));
        }
#pragma unroll
        for (int ni = 0; ni < 2; ++ni) {
            const unsigned short* bp = Bt + (size_t)(n0 + ni * 16 + ln) * K + k0;
#pragma unroll
            for (int v = 0; v < 8; ++v)
                b[ni].u[v] = *(const unsigned*)(bp + kpair(v, hf));
        }
#pragma unroll
        for (int mi = 0; mi < 2; ++mi)
#pragma unroll
            for (int ni = 0; ni < 2; ++ni)
                acc[mi][ni] = __builtin_amdgcn_wmma_f32_16x16x32_bf16(
                    false, a[mi].v, false, b[ni].v, (short)0, acc[mi][ni], false, false);
    }

#pragma unroll
    for (int mi = 0; mi < 2; ++mi)
#pragma unroll
        for (int ni = 0; ni < 2; ++ni) {
            int col = n0 + ni * 16 + ln;
            float bv = bias ? bias[col] : 0.f;
#pragma unroll
            for (int r = 0; r < 8; ++r) {
                int row = m0 + mi * 16 + r + 8 * hf;
                float v = acc[mi][ni][r] + bv;
                if (relu) v = fmaxf(v, 0.f);
                size_t o = (size_t)row * N + col;
                if (Cf) Cf[o] = v;
                if (Cb) Cb[o] = f2bf(v);
            }
        }
}

// ---------------- flash attention (one wave per head x 16-query tile) ----------
// Q,K bf16 [S,512] head-major cols; Vt bf16 [512,S]; O f32 [S,512]
__global__ void flash_attn(const unsigned short* __restrict__ Q,
                           const unsigned short* __restrict__ Km,
                           const unsigned short* __restrict__ Vt,
                           float* __restrict__ O, int S, int qtiles) {
    __shared__ unsigned short Pb[4][16][32];
    int wid = threadIdx.x >> 5;
    int lane = threadIdx.x & 31;
    int ln = lane & 15, hf = lane >> 4;
    int gw = blockIdx.x * 4 + wid;
    int h = gw / qtiles;
    int qt = gw % qtiles;
    const float scale = 0.125f;   // 1/sqrt(64)

    // Q tile fragments (K-dim 0..31, 32..63)
    Frag qa[2];
    {
        const unsigned short* qp = Q + (size_t)(qt * 16 + ln) * Dm + h * DH;
#pragma unroll
        for (int s = 0; s < 2; ++s)
#pragma unroll
            for (int v = 0; v < 8; ++v)
                qa[s].u[v] = *(const unsigned*)(qp + s * 32 + kpair(v, hf));
    }

    float mrun[8], lrun[8];
#pragma unroll
    for (int r = 0; r < 8; ++r) { mrun[r] = -3.0e38f; lrun[r] = 0.f; }
    v8f o[4];
#pragma unroll
    for (int t = 0; t < 4; ++t) o[t] = (v8f){0.f, 0.f, 0.f, 0.f, 0.f, 0.f, 0.f, 0.f};

    for (int kb0 = 0; kb0 < S; kb0 += 32) {
        // scores for two 16-key subtiles
        v8f sc[2];
#pragma unroll
        for (int t = 0; t < 2; ++t) {
            sc[t] = (v8f){0.f, 0.f, 0.f, 0.f, 0.f, 0.f, 0.f, 0.f};
            const unsigned short* kp = Km + (size_t)(kb0 + t * 16 + ln) * Dm + h * DH;
#pragma unroll
            for (int s = 0; s < 2; ++s) {
                Frag kf;
#pragma unroll
                for (int v = 0; v < 8; ++v)
                    kf.u[v] = *(const unsigned*)(kp + s * 32 + kpair(v, hf));
                sc[t] = __builtin_amdgcn_wmma_f32_16x16x32_bf16(
                    false, qa[s].v, false, kf.v, (short)0, sc[t], false, false);
            }
        }
        // online softmax, P -> LDS as bf16
#pragma unroll
        for (int r = 0; r < 8; ++r) {
            float s0 = sc[0][r] * scale, s1 = sc[1][r] * scale;
            float rm = fmaxf(s0, s1);
#pragma unroll
            for (int m = 8; m >= 1; m >>= 1) rm = fmaxf(rm, __shfl_xor(rm, m, 16));
            float mn = fmaxf(mrun[r], rm);
            float rescale = __expf(mrun[r] - mn);
            float p0 = __expf(s0 - mn), p1 = __expf(s1 - mn);
            float rs = p0 + p1;
#pragma unroll
            for (int m = 8; m >= 1; m >>= 1) rs += __shfl_xor(rs, m, 16);
            lrun[r] = lrun[r] * rescale + rs;
            mrun[r] = mn;
#pragma unroll
            for (int nt = 0; nt < 4; ++nt) o[nt][r] *= rescale;
            int prow = r + 8 * hf;
            Pb[wid][prow][ln]      = f2bf(p0);
            Pb[wid][prow][16 + ln] = f2bf(p1);
        }
        asm volatile("s_wait_dscnt 0" ::: "memory");
        // P as A-fragment [16 x 32]
        Frag pa;
#pragma unroll
        for (int v = 0; v < 8; ++v)
            pa.u[v] = *(const unsigned*)(&Pb[wid][ln][kpair(v, hf)]);
        // O += P @ V
#pragma unroll
        for (int nt = 0; nt < 4; ++nt) {
            Frag vb;
            const unsigned short* vp = Vt + (size_t)(h * DH + nt * 16 + ln) * S + kb0;
#pragma unroll
            for (int v = 0; v < 8; ++v)
                vb.u[v] = *(const unsigned*)(vp + kpair(v, hf));
            o[nt] = __builtin_amdgcn_wmma_f32_16x16x32_bf16(
                false, pa.v, false, vb.v, (short)0, o[nt], false, false);
        }
    }
    // normalize and store
#pragma unroll
    for (int nt = 0; nt < 4; ++nt)
#pragma unroll
        for (int r = 0; r < 8; ++r) {
            int row = qt * 16 + r + 8 * hf;
            int col = h * DH + nt * 16 + ln;
            O[(size_t)row * Dm + col] = o[nt][r] / lrun[r];
        }
}

// ---------------- host orchestration ----------------
static inline dim3 g1(int n, int b = 256) { return dim3((n + b - 1) / b); }

extern "C" void kernel_launch(void* const* d_in, const int* in_sizes, int n_in,
                              void* d_out, int out_size, void* d_ws, size_t ws_size,
                              hipStream_t stream) {
    (void)in_sizes; (void)n_in; (void)out_size; (void)ws_size;
    const float* out_atom  = (const float*)d_in[0];
    const float* out_fasta = (const float*)d_in[1];
    const int*   query     = (const int*)d_in[2];
    const int*   cent      = (const int*)d_in[3];
    const float* a_in_w  = (const float*)d_in[4];  const float* a_in_b  = (const float*)d_in[5];
    const float* a_out_w = (const float*)d_in[6];  const float* a_out_b = (const float*)d_in[7];
    const float* an1_g   = (const float*)d_in[8];  const float* an1_b   = (const float*)d_in[9];
    const float* fw1_w1  = (const float*)d_in[10]; const float* fw1_b1  = (const float*)d_in[11];
    const float* fw1_w2  = (const float*)d_in[12]; const float* fw1_b2  = (const float*)d_in[13];
    const float* an2_g   = (const float*)d_in[14]; const float* an2_b   = (const float*)d_in[15];
    const float* fw2_w1  = (const float*)d_in[16]; const float* fw2_b1  = (const float*)d_in[17];
    const float* fw2_w2  = (const float*)d_in[18]; const float* fw2_b2  = (const float*)d_in[19];
    const float* f_in_w  = (const float*)d_in[20]; const float* f_in_b  = (const float*)d_in[21];
    const float* f_out_w = (const float*)d_in[22]; const float* f_out_b = (const float*)d_in[23];
    const float* ans2_g  = (const float*)d_in[24]; const float* ans2_b  = (const float*)d_in[25];
    const float* lrl_w1  = (const float*)d_in[26]; const float* lrl_b1  = (const float*)d_in[27];
    const float* lrl_w2  = (const float*)d_in[28]; const float* lrl_b2  = (const float*)d_in[29];

    size_t off = 0;
    auto alloc = [&](size_t bytes) -> char* {
        char* p = (char*)d_ws + off;
        off += (bytes + 255) & ~(size_t)255;
        return p;
    };
    int*            resmap   = (int*)alloc(NAa * 4);
    float*          atomin_f = (float*)alloc((size_t)NAa * Dm * 4);
    unsigned short* atomin_b = (unsigned short*)alloc((size_t)NAa * Dm * 2);
    unsigned short* w_ain_t  = (unsigned short*)alloc((size_t)NQKV * Dm * 2);
    float*          qkv_f    = (float*)alloc((size_t)NAa * NQKV * 4);
    unsigned short* q_b      = (unsigned short*)alloc((size_t)NAa * Dm * 2);
    unsigned short* k_b      = (unsigned short*)alloc((size_t)NAa * Dm * 2);
    unsigned short* vt_b     = (unsigned short*)alloc((size_t)Dm * NAa * 2);
    float*          ao_f     = (float*)alloc((size_t)NAa * Dm * 4);
    unsigned short* ao_b     = (unsigned short*)alloc((size_t)NAa * Dm * 2);
    unsigned short* w_aout_t = (unsigned short*)alloc((size_t)Dm * Dm * 2);
    float*          proj_f   = (float*)alloc((size_t)NAa * Dm * 4);
    float*          h1_f     = (float*)alloc((size_t)NAa * Dm * 4);
    unsigned short* h1_b     = (unsigned short*)alloc((size_t)NAa * Dm * 2);
    unsigned short* w_f11_t  = (unsigned short*)alloc((size_t)F1 * Dm * 2);
    unsigned short* mid1_b   = (unsigned short*)alloc((size_t)NAa * F1 * 2);
    unsigned short* w_f12_t  = (unsigned short*)alloc((size_t)Dm * F1 * 2);
    float*          atr_f    = (float*)alloc((size_t)NAa * Dm * 4);
    unsigned short* h2_b     = (unsigned short*)alloc((size_t)NAa * Dm * 2);
    unsigned short* w_f21_t  = (unsigned short*)alloc((size_t)F2 * Dm * 2);
    unsigned short* mid2_b   = (unsigned short*)alloc((size_t)NAa * F2 * 2);
    unsigned short* w_f22_t  = (unsigned short*)alloc((size_t)Dm * F2 * 2);
    float*          oal_f    = (float*)alloc((size_t)NFf * Dm * 4);
    float*          fq_f     = (float*)alloc((size_t)NFf * Dm * 4);
    unsigned short* fq_b     = (unsigned short*)alloc((size_t)NFf * Dm * 2);
    unsigned short* w_fin_t  = (unsigned short*)alloc((size_t)NQKV * Dm * 2);
    float*          fqkv_f   = (float*)alloc((size_t)NFf * NQKV * 4);
    unsigned short* fqb      = (unsigned short*)alloc((size_t)NFf * Dm * 2);
    unsigned short* fkb      = (unsigned short*)alloc((size_t)NFf * Dm * 2);
    unsigned short* fvtb     = (unsigned short*)alloc((size_t)Dm * NFf * 2);
    float*          fao_f    = (float*)alloc((size_t)NFf * Dm * 4);
    unsigned short* fao_b    = (unsigned short*)alloc((size_t)NFf * Dm * 2);
    unsigned short* w_fout_t = (unsigned short*)alloc((size_t)Dm * Dm * 2);
    float*          fproj_f  = (float*)alloc((size_t)NFf * Dm * 4);
    unsigned short* ft_b     = (unsigned short*)alloc((size_t)NFf * Dm * 2);
    unsigned short* w_l1_t   = (unsigned short*)alloc((size_t)F2 * Dm * 2);
    unsigned short* fmid_b   = (unsigned short*)alloc((size_t)NFf * F2 * 2);
    unsigned short* w_l2_t   = (unsigned short*)alloc((size_t)Dm * F2 * 2);

    float* atom_out_d  = (float*)d_out;
    float* fasta_out_d = (float*)d_out + (size_t)NAa * Dm;

    const dim3 B256(256), B128(128);

    // weights -> bf16 [N,K]
    tconv<<<g1(Dm * NQKV), B256, 0, stream>>>(a_in_w,  w_ain_t,  Dm, NQKV);
    tconv<<<g1(Dm * Dm),   B256, 0, stream>>>(a_out_w, w_aout_t, Dm, Dm);
    tconv<<<g1(Dm * F1),   B256, 0, stream>>>(fw1_w1,  w_f11_t,  Dm, F1);
    tconv<<<g1(F1 * Dm),   B256, 0, stream>>>(fw1_w2,  w_f12_t,  F1, Dm);
    tconv<<<g1(Dm * F2),   B256, 0, stream>>>(fw2_w1,  w_f21_t,  Dm, F2);
    tconv<<<g1(F2 * Dm),   B256, 0, stream>>>(fw2_w2,  w_f22_t,  F2, Dm);
    tconv<<<g1(Dm * NQKV), B256, 0, stream>>>(f_in_w,  w_fin_t,  Dm, NQKV);
    tconv<<<g1(Dm * Dm),   B256, 0, stream>>>(f_out_w, w_fout_t, Dm, Dm);
    tconv<<<g1(Dm * F2),   B256, 0, stream>>>(lrl_w1,  w_l1_t,   Dm, F2);
    tconv<<<g1(F2 * Dm),   B256, 0, stream>>>(lrl_w2,  w_l2_t,   F2, Dm);

    // atom_in
    build_resmap<<<1, NFf, 0, stream>>>(query, resmap);
    repeat_add<<<g1(NAa * Dm), B256, 0, stream>>>(out_atom, out_fasta, resmap, atomin_f, atomin_b);

    // atom attention
    gemm_wmma_bf16<<<dim3(NQKV / 64, NAa / 64), B128, 0, stream>>>(
        atomin_b, w_ain_t, a_in_b, qkv_f, nullptr, NAa, NQKV, Dm, 0);
    qkv_split<<<g1(NAa * Dm), B256, 0, stream>>>(qkv_f, q_b, k_b, vt_b, NAa);
    flash_attn<<<dim3(Hh * (NAa / 16) / 4), B128, 0, stream>>>(q_b, k_b, vt_b, ao_f, NAa, NAa / 16);
    conv_bf16<<<g1(NAa * Dm), B256, 0, stream>>>(ao_f, ao_b, NAa * Dm);
    gemm_wmma_bf16<<<dim3(Dm / 64, NAa / 64), B128, 0, stream>>>(
        ao_b, w_aout_t, a_out_b, proj_f, nullptr, NAa, Dm, Dm, 0);
    ln512<<<dim3(NAa), B256, 0, stream>>>(atomin_f, proj_f, an1_g, an1_b, h1_f, h1_b);

    // FFN1 -> atom_trans
    gemm_wmma_bf16<<<dim3(F1 / 64, NAa / 64), B128, 0, stream>>>(
        h1_b, w_f11_t, fw1_b1, nullptr, mid1_b, NAa, F1, Dm, 1);
    gemm_wmma_bf16<<<dim3(Dm / 64, NAa / 64), B128, 0, stream>>>(
        mid1_b, w_f12_t, fw1_b2, atr_f, nullptr, NAa, Dm, F1, 0);
    ln512<<<dim3(NAa), B256, 0, stream>>>(h1_f, atr_f, an2_g, an2_b, nullptr, h2_b);

    // FFN2 -> atom_out (output 0)
    gemm_wmma_bf16<<<dim3(F2 / 64, NAa / 64), B128, 0, stream>>>(
        h2_b, w_f21_t, fw2_b1, nullptr, mid2_b, NAa, F2, Dm, 1);
    gemm_wmma_bf16<<<dim3(Dm / 64, NAa / 64), B128, 0, stream>>>(
        mid2_b, w_f22_t, fw2_b2, atom_out_d, nullptr, NAa, Dm, F2, 0);

    // centroid gather -> fasta_q
    gather_max_sum<<<g1(NFf * Dm), B256, 0, stream>>>(atr_f, cent, oal_f);
    add2<<<g1(NFf * Dm), B256, 0, stream>>>(out_fasta, oal_f, fq_f, fq_b, NFf * Dm);

    // fasta attention
    gemm_wmma_bf16<<<dim3(NQKV / 64, NFf / 64), B128, 0, stream>>>(
        fq_b, w_fin_t, f_in_b, fqkv_f, nullptr, NFf, NQKV, Dm, 0);
    qkv_split<<<g1(NFf * Dm), B256, 0, stream>>>(fqkv_f, fqb, fkb, fvtb, NFf);
    flash_attn<<<dim3(Hh * (NFf / 16) / 4), B128, 0, stream>>>(fqb, fkb, fvtb, fao_f, NFf, NFf / 16);
    conv_bf16<<<g1(NFf * Dm), B256, 0, stream>>>(fao_f, fao_b, NFf * Dm);
    gemm_wmma_bf16<<<dim3(Dm / 64, NFf / 64), B128, 0, stream>>>(
        fao_b, w_fout_t, f_out_b, fproj_f, nullptr, NFf, Dm, Dm, 0);
    ln512<<<dim3(NFf), B256, 0, stream>>>(fq_f, fproj_f, ans2_g, ans2_b, nullptr, ft_b);

    // fasta FFN -> fasta_trans (output 1)
    gemm_wmma_bf16<<<dim3(F2 / 64, NFf / 64), B128, 0, stream>>>(
        ft_b, w_l1_t, lrl_b1, nullptr, fmid_b, NFf, F2, Dm, 1);
    gemm_wmma_bf16<<<dim3(Dm / 64, NFf / 64), B128, 0, stream>>>(
        fmid_b, w_l2_t, lrl_b2, fasta_out_d, nullptr, NFf, Dm, F2, 0);
}
